// CrossAttention_3985729650899
// MI455X (gfx1250) — compile-verified, matrix operand order checked by reading, not run on hardware
//
#include <hip/hip_runtime.h>
#include <hip/hip_bf16.h>

typedef __attribute__((ext_vector_type(16))) __bf16 v16bf;
typedef __attribute__((ext_vector_type(8)))  __bf16 v8bf;
typedef __attribute__((ext_vector_type(8)))  float  v8f;
typedef __attribute__((ext_vector_type(4)))  unsigned int v4u;
typedef __attribute__((ext_vector_type(8)))  int v8i;
typedef __attribute__((ext_vector_type(4)))  int v4i;

#define N_HEADS   16
#define DHEAD     64
#define BATCH     16
#define NQ        2048
#define DMODEL    1024
#define LCTX      333
#define TXT_LEN   77
#define LTXT_PAD  96
#define LIMG      256
#define QK_SCALE  0.125f   // 64^-0.5

// ---- WMMA helpers ---------------------------------------------------------
static __device__ __forceinline__ v8f wmma_bf16(v16bf a, v16bf b, v8f c) {
  return __builtin_amdgcn_wmma_f32_16x16x32_bf16(
      false, a, false, b, (short)0, c, false, false);
}

// Fragment for 16x16x32 bf16: per-lane 8 contiguous bf16 at p (K=half*8..+7)
// and 8 more at p+16 (K=16+half*8..+7).
static __device__ __forceinline__ v16bf load_frag(const __bf16* p) {
  v8bf lo = *(const v8bf*)(p);
  v8bf hi = *(const v8bf*)(p + 16);
  v16bf r;
#pragma unroll
  for (int i = 0; i < 8; ++i) { r[i] = lo[i]; r[i + 8] = hi[i]; }
  return r;
}

// ---- TDM: 2-D tensor tile load (global -> LDS), element size 2B -----------
static __device__ __forceinline__ void tdm_load_2d(
    unsigned lds_addr, const void* gptr, unsigned tensor_d0, unsigned tensor_d1,
    unsigned tile_d0, unsigned tile_d1, unsigned stride0_elems) {
  unsigned long long ga = (unsigned long long)(size_t)gptr;
  v4u g0;
  g0[0] = 1u;                                         // count=1 (user D#)
  g0[1] = lds_addr;                                   // lds_addr (bytes)
  g0[2] = (unsigned)(ga & 0xffffffffu);               // global_addr[31:0]
  g0[3] = (unsigned)((ga >> 32) & 0x01ffffffu)        // global_addr[56:32]
        | (2u << 30);                                 // type=2 ("image")
  v8i g1;
  g1[0] = (int)(1u << 16);                            // data_size=1 (2 bytes)
  g1[1] = (int)((tensor_d0 & 0xffffu) << 16);         // tensor_dim0[15:0]
  g1[2] = (int)((tensor_d0 >> 16) | ((tensor_d1 & 0xffffu) << 16));
  g1[3] = (int)((tensor_d1 >> 16) | (tile_d0 << 16)); // tile_dim0
  g1[4] = (int)(tile_d1 & 0xffffu);                   // tile_dim1 (tile_dim2=0)
  g1[5] = (int)stride0_elems;                         // tensor_dim0_stride lo
  g1[6] = 0;                                          // stride0 hi / stride1 lo
  g1[7] = 0;
  v4i z4 = {0, 0, 0, 0};
#if __clang_major__ >= 23
  v8i z8 = {0, 0, 0, 0, 0, 0, 0, 0};
  __builtin_amdgcn_tensor_load_to_lds(g0, g1, z4, z4, z8, 0);
#else
  __builtin_amdgcn_tensor_load_to_lds(g0, g1, z4, z4, 0);
#endif
}

// ---- Conversion kernels ---------------------------------------------------
__global__ void ca_cvt_bf16(const float* __restrict__ in,
                            __bf16* __restrict__ out, int n) {
  int i = blockIdx.x * 256 + threadIdx.x;
  if (i < n) out[i] = (__bf16)in[i];
}

__global__ void ca_transpose_cvt(const float* __restrict__ W,
                                 __bf16* __restrict__ Wt) {
  int idx = blockIdx.x * 256 + threadIdx.x;   // 1024*1024 threads
  int k = idx >> 10, n = idx & 1023;
  Wt[(size_t)n * DMODEL + k] = (__bf16)W[idx];
}

// ---- Projection GEMM: 32x32 tile per wave (2x2 WMMA grid) -----------------
__global__ __launch_bounds__(256) void ca_gemm_proj(
    const __bf16* __restrict__ A, const __bf16* __restrict__ Bt,
    __bf16* __restrict__ C, int mtiles_per_batch, int src_bstride,
    int src_off, int dst_bstride, float scale, int tiles_total) {
  int wave = threadIdx.x >> 5, lane = threadIdx.x & 31;
  int half = lane >> 4, l16 = lane & 15;
  int tile = blockIdx.x * 8 + wave;
  if (tile >= tiles_total) return;                    // wave-uniform
  int nt = tile & 31, mt = tile >> 5;
  int b  = mt / mtiles_per_batch;
  int r0 = (mt - b * mtiles_per_batch) << 5;
  int n0 = nt << 5;
  const __bf16* a0 = A + (size_t)(b * src_bstride + src_off + r0 + l16) * DMODEL;
  const __bf16* a1 = a0 + (size_t)16 * DMODEL;
  const __bf16* b0 = Bt + (size_t)(n0 + l16) * DMODEL;
  const __bf16* b1 = b0 + (size_t)16 * DMODEL;
  v8f acc00 = {}, acc01 = {}, acc10 = {}, acc11 = {};
#pragma unroll 2
  for (int kk = 0; kk < DMODEL; kk += 32) {
    int ko = kk + half * 8;
    __builtin_prefetch(a0 + kk + 512, 0, 1);          // global_prefetch_b8
    v16bf A0 = load_frag(a0 + ko), A1 = load_frag(a1 + ko);
    v16bf B0 = load_frag(b0 + ko), B1 = load_frag(b1 + ko);
    acc00 = wmma_bf16(A0, B0, acc00);
    acc01 = wmma_bf16(A0, B1, acc01);
    acc10 = wmma_bf16(A1, B0, acc10);
    acc11 = wmma_bf16(A1, B1, acc11);
  }
  size_t base = (size_t)(b * dst_bstride + r0 + half * 8) * DMODEL + n0 + l16;
#pragma unroll
  for (int r = 0; r < 8; ++r) {
    C[base + (size_t)r * DMODEL]                       = (__bf16)(acc00[r] * scale);
    C[base + (size_t)r * DMODEL + 16]                  = (__bf16)(acc01[r] * scale);
    C[base + (size_t)(r + 16) * DMODEL]                = (__bf16)(acc10[r] * scale);
    C[base + (size_t)(r + 16) * DMODEL + 16]           = (__bf16)(acc11[r] * scale);
  }
}

// ---- Output GEMM: 32x32 tile, fp32 out + bias -----------------------------
__global__ __launch_bounds__(256) void ca_gemm_out(
    const __bf16* __restrict__ A, const __bf16* __restrict__ Bt,
    const float* __restrict__ bias, float* __restrict__ C, int tiles_total) {
  int wave = threadIdx.x >> 5, lane = threadIdx.x & 31;
  int half = lane >> 4, l16 = lane & 15;
  int tile = blockIdx.x * 8 + wave;
  if (tile >= tiles_total) return;
  int nt = tile & 31, mt = tile >> 5;
  int r0 = mt << 5, n0 = nt << 5;
  const __bf16* a0 = A + (size_t)(r0 + l16) * DMODEL;
  const __bf16* a1 = a0 + (size_t)16 * DMODEL;
  const __bf16* b0 = Bt + (size_t)(n0 + l16) * DMODEL;
  const __bf16* b1 = b0 + (size_t)16 * DMODEL;
  v8f acc00 = {}, acc01 = {}, acc10 = {}, acc11 = {};
#pragma unroll 2
  for (int kk = 0; kk < DMODEL; kk += 32) {
    int ko = kk + half * 8;
    __builtin_prefetch(a0 + kk + 512, 0, 1);
    v16bf A0 = load_frag(a0 + ko), A1 = load_frag(a1 + ko);
    v16bf B0 = load_frag(b0 + ko), B1 = load_frag(b1 + ko);
    acc00 = wmma_bf16(A0, B0, acc00);
    acc01 = wmma_bf16(A0, B1, acc01);
    acc10 = wmma_bf16(A1, B0, acc10);
    acc11 = wmma_bf16(A1, B1, acc11);
  }
  float bn0 = bias[n0 + l16], bn1 = bias[n0 + 16 + l16];
  size_t base = (size_t)(r0 + half * 8) * DMODEL + n0 + l16;
#pragma unroll
  for (int r = 0; r < 8; ++r) {
    C[base + (size_t)r * DMODEL]             = acc00[r] + bn0;
    C[base + (size_t)r * DMODEL + 16]        = acc01[r] + bn1;
    C[base + (size_t)(r + 16) * DMODEL]      = acc10[r] + bn0;
    C[base + (size_t)(r + 16) * DMODEL + 16] = acc11[r] + bn1;
  }
}

// ---- Attention ------------------------------------------------------------
// One segment, flash-style online softmax over LPAD keys (NCH chunks of 32).
template <int LPAD, int NCH>
static __device__ __forceinline__ void ca_attn_seg(
    const __bf16* __restrict__ kseg, const __bf16* __restrict__ vseg,
    int Lvalid, int b, int h, const v16bf* Qf, v8f* Oacc,
    __bf16* sV, __bf16* sVT, __bf16* pw, int tid, int half, int l16) {
  __syncthreads();
  // TDM: async DMA of the V tile (LPAD x 64, row stride DMODEL) into LDS.
  if (tid < 32) {
    tdm_load_2d((unsigned)(size_t)(void*)sV,
                vseg + (size_t)b * LPAD * DMODEL + h * DHEAD,
                /*tensor_d0=*/DHEAD, /*tensor_d1=*/LPAD,
                /*tile_d0=*/DHEAD, /*tile_d1=*/LPAD,
                /*stride0=*/DMODEL);
    __builtin_amdgcn_s_wait_tensorcnt(0);
  }
  __syncthreads();
  // LDS->LDS transpose: sVT[d*256 + j] = sV[j*64 + d]
  for (int idx = tid; idx < DHEAD * LPAD; idx += 256) {
    int d = idx & 63, j = idx >> 6;
    sVT[d * 256 + j] = sV[j * DHEAD + d];
  }
  __syncthreads();

  float mrun[8], lrun[8];
  v8f O[4] = {};
#pragma unroll
  for (int r = 0; r < 8; ++r) { mrun[r] = -3.0e38f; lrun[r] = 0.0f; }

  for (int jc = 0; jc < NCH; ++jc) {
    int j0 = jc * 32;
    v8f S0 = {}, S1 = {};
    const __bf16* kb = kseg + (size_t)(b * LPAD + j0) * DMODEL + h * DHEAD;
#pragma unroll
    for (int ks = 0; ks < 2; ++ks) {
      v16bf B0 = load_frag(kb + (size_t)l16 * DMODEL + ks * 32 + half * 8);
      v16bf B1 = load_frag(kb + (size_t)(16 + l16) * DMODEL + ks * 32 + half * 8);
      S0 = wmma_bf16(Qf[ks], B0, S0);
      S1 = wmma_bf16(Qf[ks], B1, S1);
    }
    if (j0 + l16 >= Lvalid) {
#pragma unroll
      for (int r = 0; r < 8; ++r) S0[r] = -1.0e30f;
    }
    if (j0 + 16 + l16 >= Lvalid) {
#pragma unroll
      for (int r = 0; r < 8; ++r) S1[r] = -1.0e30f;
    }
    float rmax[8], p0[8], p1[8], rsum[8];
#pragma unroll
    for (int r = 0; r < 8; ++r) rmax[r] = fmaxf(S0[r], S1[r]);
#pragma unroll
    for (int off = 1; off < 16; off <<= 1)
#pragma unroll
      for (int r = 0; r < 8; ++r)
        rmax[r] = fmaxf(rmax[r], __shfl_xor(rmax[r], off, 32));
#pragma unroll
    for (int r = 0; r < 8; ++r) {
      float mn = fmaxf(mrun[r], rmax[r]);
      float al = __expf(mrun[r] - mn);
      mrun[r] = mn;
      p0[r] = __expf(S0[r] - mn);
      p1[r] = __expf(S1[r] - mn);
      rsum[r] = p0[r] + p1[r];
      lrun[r] *= al;
#pragma unroll
      for (int t = 0; t < 4; ++t) O[t][r] *= al;
    }
#pragma unroll
    for (int off = 1; off < 16; off <<= 1)
#pragma unroll
      for (int r = 0; r < 8; ++r)
        rsum[r] += __shfl_xor(rsum[r], off, 32);
#pragma unroll
    for (int r = 0; r < 8; ++r) lrun[r] += rsum[r];

    // D-layout -> A-layout bounce of P through per-wave LDS (16x32 bf16)
#pragma unroll
    for (int r = 0; r < 8; ++r) {
      int m = r + 8 * half;
      pw[m * 32 + l16]      = (__bf16)p0[r];
      pw[m * 32 + 16 + l16] = (__bf16)p1[r];
    }
    asm volatile("s_wait_dscnt 0x0" ::: "memory");
    v16bf Pf = load_frag(pw + l16 * 32 + half * 8);
#pragma unroll
    for (int t = 0; t < 4; ++t) {
      v16bf Vf = load_frag(sVT + (size_t)(t * 16 + l16) * 256 + j0 + half * 8);
      O[t] = wmma_bf16(Pf, Vf, O[t]);
    }
  }
#pragma unroll
  for (int r = 0; r < 8; ++r) {
    float inv = 1.0f / lrun[r];
#pragma unroll
    for (int t = 0; t < 4; ++t) Oacc[t][r] += O[t][r] * inv;
  }
}

__global__ __launch_bounds__(256) void ca_attn(
    const __bf16* __restrict__ q,
    const __bf16* __restrict__ ktxt, const __bf16* __restrict__ vtxt,
    const __bf16* __restrict__ kimg, const __bf16* __restrict__ vimg,
    __bf16* __restrict__ out) {
  __shared__ __bf16 sV[LIMG * DHEAD];    // TDM landing tile, row-major (32 KB)
  __shared__ __bf16 sVT[DHEAD * 256];    // transposed V (32 KB)
  __shared__ __bf16 sP[8 * 16 * 32];     // per-wave P bounce (8 KB)
  int tid = threadIdx.x;
  int wave = tid >> 5, lane = tid & 31, half = lane >> 4, l16 = lane & 15;
  int b = blockIdx.z, h = blockIdx.y;
  int q0 = blockIdx.x * 128 + wave * 16;

  const __bf16* qbase =
      q + (size_t)(b * NQ + q0 + l16) * DMODEL + h * DHEAD;
  v16bf Qf[2];
#pragma unroll
  for (int ks = 0; ks < 2; ++ks) Qf[ks] = load_frag(qbase + ks * 32 + half * 8);

  __bf16* pw = sP + wave * 512;
  v8f Oacc[4] = {};

  ca_attn_seg<LTXT_PAD, 3>(ktxt, vtxt, TXT_LEN, b, h, Qf, Oacc, sV, sVT, pw,
                           tid, half, l16);
  ca_attn_seg<LIMG, 8>(kimg, vimg, LIMG, b, h, Qf, Oacc, sV, sVT, pw,
                       tid, half, l16);

  size_t orow = (size_t)(b * NQ + q0 + half * 8) * DMODEL + h * DHEAD + l16;
#pragma unroll
  for (int t = 0; t < 4; ++t)
#pragma unroll
    for (int r = 0; r < 8; ++r)
      out[orow + (size_t)r * DMODEL + t * 16] = (__bf16)Oacc[t][r];
}

// ---- Host-side orchestration ----------------------------------------------
extern "C" void kernel_launch(void* const* d_in, const int* in_sizes, int n_in,
                              void* d_out, int out_size, void* d_ws, size_t ws_size,
                              hipStream_t stream) {
  const float* x    = (const float*)d_in[0];
  const float* ctx  = (const float*)d_in[1];
  const float* Wq   = (const float*)d_in[2];
  const float* Wk   = (const float*)d_in[3];
  const float* Wv   = (const float*)d_in[4];
  const float* Wkip = (const float*)d_in[5];
  const float* Wvip = (const float*)d_in[6];
  const float* Wo   = (const float*)d_in[7];
  const float* bo   = (const float*)d_in[8];
  float* out = (float*)d_out;

  char* ws = (char*)d_ws;
  size_t off = 0;
  auto carve = [&](size_t elems) -> __bf16* {
    __bf16* p = (__bf16*)(ws + off);
    off += (elems * sizeof(__bf16) + 255) & ~(size_t)255;
    return p;
  };
  const size_t NX = (size_t)BATCH * NQ * DMODEL;
  const size_t NC = (size_t)BATCH * LCTX * DMODEL;
  const size_t NW = (size_t)DMODEL * DMODEL;
  __bf16* xbf   = carve(NX);
  __bf16* cbf   = carve(NC);
  __bf16* WqT   = carve(NW);
  __bf16* WkT   = carve(NW);
  __bf16* WvT   = carve(NW);
  __bf16* WkipT = carve(NW);
  __bf16* WvipT = carve(NW);
  __bf16* WoT   = carve(NW);
  __bf16* qbf   = carve(NX);
  __bf16* kt    = carve((size_t)BATCH * LTXT_PAD * DMODEL);
  __bf16* vt    = carve((size_t)BATCH * LTXT_PAD * DMODEL);
  __bf16* ki    = carve((size_t)BATCH * LIMG * DMODEL);
  __bf16* vi    = carve((size_t)BATCH * LIMG * DMODEL);
  __bf16* abf   = carve(NX);

  // 1) conversions
  ca_cvt_bf16<<<(int)((NX + 255) / 256), 256, 0, stream>>>(x, xbf, (int)NX);
  ca_cvt_bf16<<<(int)((NC + 255) / 256), 256, 0, stream>>>(ctx, cbf, (int)NC);
  ca_transpose_cvt<<<4096, 256, 0, stream>>>(Wq, WqT);
  ca_transpose_cvt<<<4096, 256, 0, stream>>>(Wk, WkT);
  ca_transpose_cvt<<<4096, 256, 0, stream>>>(Wv, WvT);
  ca_transpose_cvt<<<4096, 256, 0, stream>>>(Wkip, WkipT);
  ca_transpose_cvt<<<4096, 256, 0, stream>>>(Wvip, WvipT);
  ca_transpose_cvt<<<4096, 256, 0, stream>>>(Wo, WoT);

  // 2) projections (32-row x 32-col tiles per wave; q pre-scaled)
  // q: (16*2048 x 1024) -> 64 mtiles/batch * 32 ntiles * 16 batches = 32768
  ca_gemm_proj<<<32768 / 8, 256, 0, stream>>>(
      xbf, WqT, qbf, 64, NQ, 0, NQ, QK_SCALE, 32768);
  // k,v text: 3 mtiles/batch (96 rows, 77 valid) -> 1536 tiles
  ca_gemm_proj<<<1536 / 8, 256, 0, stream>>>(
      cbf, WkT, kt, 3, LCTX, 0, LTXT_PAD, 1.0f, 1536);
  ca_gemm_proj<<<1536 / 8, 256, 0, stream>>>(
      cbf, WvT, vt, 3, LCTX, 0, LTXT_PAD, 1.0f, 1536);
  // k,v image: 8 mtiles/batch (256 rows) -> 4096 tiles
  ca_gemm_proj<<<4096 / 8, 256, 0, stream>>>(
      cbf, WkipT, ki, 8, LCTX, TXT_LEN, LIMG, 1.0f, 4096);
  ca_gemm_proj<<<4096 / 8, 256, 0, stream>>>(
      cbf, WvipT, vi, 8, LCTX, TXT_LEN, LIMG, 1.0f, 4096);

  // 3) dual-segment attention (TDM-staged V, WMMA QK^T and PV)
  dim3 agrid(NQ / 128, N_HEADS, BATCH);
  ca_attn<<<agrid, 256, 0, stream>>>(qbf, kt, vt, ki, vi, abf);

  // 4) output projection + bias -> fp32
  ca_gemm_out<<<32768 / 8, 256, 0, stream>>>(abf, WoT, bo, out, 32768);
}
